// OrthFusion_7327214207194
// MI455X (gfx1250) — matrix operand order repaired and checked by reading, not compiled, not run on hardware
//
#include <hip/hip_runtime.h>
#include <hip/hip_bf16.h>

// Problem constants (from reference)
#define IND0   1024
#define IND1   1024
#define KDIM   1024
#define OUTD   128
#define BATCHN 65536

#define KC     128      // K chunk staged in LDS per iteration
#define MTILE  16       // batch rows per block

typedef __attribute__((ext_vector_type(16))) __bf16 v16bf;
typedef __attribute__((ext_vector_type(8)))  __bf16 v8bf;
typedef __attribute__((ext_vector_type(4)))  __bf16 v4bf;
typedef __attribute__((ext_vector_type(8)))  float  v8f;

// ---------------------------------------------------------------------------
// Prep: OWN-norm each weight row (mean-center + L2 normalize), store bf16
// row-major into workspace:  Wn[0..127]  = W0n rows, Wn[128..255] = W1n rows.
// One block per (matrix, row): grid = 256, block = 256.
// ---------------------------------------------------------------------------
__global__ __launch_bounds__(256)
void own_norm_kernel(const float* __restrict__ W,   // [128, 2048]
                     __bf16* __restrict__ Wn)       // [256, 1024]
{
    const int row = blockIdx.x & 127;
    const int mat = blockIdx.x >> 7;
    const float* src = W + row * (IND0 + IND1) + mat * IND0;

    __shared__ float red[256];
    const int t = threadIdx.x;

    float v0 = src[t], v1 = src[t + 256], v2 = src[t + 512], v3 = src[t + 768];

    // block reduce: sum
    red[t] = v0 + v1 + v2 + v3;
    __syncthreads();
    for (int off = 128; off > 0; off >>= 1) {
        if (t < off) red[t] += red[t + off];
        __syncthreads();
    }
    const float mean = red[0] * (1.0f / 1024.0f);
    __syncthreads();

    v0 -= mean; v1 -= mean; v2 -= mean; v3 -= mean;

    // block reduce: sum of squares of centered row
    red[t] = v0 * v0 + v1 * v1 + v2 * v2 + v3 * v3;
    __syncthreads();
    for (int off = 128; off > 0; off >>= 1) {
        if (t < off) red[t] += red[t + off];
        __syncthreads();
    }
    const float rs = rsqrtf(red[0]);

    __bf16* dst = Wn + (mat * 128 + row) * KDIM;
    dst[t]       = (__bf16)(v0 * rs);
    dst[t + 256] = (__bf16)(v1 * rs);
    dst[t + 512] = (__bf16)(v2 * rs);
    dst[t + 768] = (__bf16)(v3 * rs);
}

// ---------------------------------------------------------------------------
// Main fused GEMM: out0 = x @ W0n.T, out1 = y @ W1n.T,
//                  outf = out0*a0 + out1*a1 + bias
// Block = 256 threads = 8 wave32s. Block owns MTILE=16 batch rows; wave w
// owns output columns [16w, 16w+16). K staged through LDS in KC chunks,
// converted f32->bf16 ONCE at staging time (shared by all 8 waves).
// v_wmma_f32_16x16x32_bf16 per 32-wide K step, two accumulators per wave.
// ---------------------------------------------------------------------------
__global__ __launch_bounds__(256)
void fused_ownnorm_gemm_kernel(const float* __restrict__ x,
                               const float* __restrict__ y,
                               const __bf16* __restrict__ Wn,   // [256,1024] bf16
                               const float* __restrict__ bias,
                               const float* __restrict__ a0,
                               const float* __restrict__ a1,
                               float* __restrict__ out0,
                               float* __restrict__ out1,
                               float* __restrict__ outf)
{
    __shared__ __bf16 ldsx[MTILE * KC];   // 4 KB
    __shared__ __bf16 ldsy[MTILE * KC];   // 4 KB

    const int tid   = threadIdx.x;
    const int wave  = tid >> 5;          // 0..7 -> N tile
    const int lane  = tid & 31;
    const int lsel  = lane & 15;
    const int half  = lane >> 4;         // 0: lanes 0-15, 1: lanes 16-31
    const int mrow0 = blockIdx.x * MTILE;

    // B-fragment source rows (row-major bf16 weights; lane lsel carries
    // output column wave*16+lsel, two 8-bf16 K runs, 16 apart, per fragment)
    const __bf16* w0row = Wn + (wave * 16 + lsel) * KDIM;
    const __bf16* w1row = Wn + (128 + wave * 16 + lsel) * KDIM;

    v8f acc0 = {};   // x @ W0n.T accumulator (16x16 f32 tile)
    v8f acc1 = {};   // y @ W1n.T accumulator

    for (int kc = 0; kc < KDIM; kc += KC) {
        __syncthreads();   // protect LDS WAR vs. previous chunk's reads

        // Stage 16 x KC elements of x and y into LDS as bf16 (convert once):
        // 512 float4 loads each, 2 per thread; 4-bf16 (8 B) LDS stores.
        for (int i = tid; i < MTILE * (KC / 4); i += 256) {
            const int r  = i / (KC / 4);
            const int c4 = i % (KC / 4);
            const float4 vx =
                ((const float4*)(x + (size_t)(mrow0 + r) * KDIM + kc))[c4];
            const float4 vy =
                ((const float4*)(y + (size_t)(mrow0 + r) * KDIM + kc))[c4];
            v4bf px, py;
            px[0] = (__bf16)vx.x; px[1] = (__bf16)vx.y;
            px[2] = (__bf16)vx.z; px[3] = (__bf16)vx.w;
            py[0] = (__bf16)vy.x; py[1] = (__bf16)vy.y;
            py[2] = (__bf16)vy.z; py[3] = (__bf16)vy.w;
            *(v4bf*)(ldsx + r * KC + c4 * 4) = px;
            *(v4bf*)(ldsy + r * KC + c4 * 4) = py;
        }
        __syncthreads();

        for (int ks = 0; ks < KC; ks += 32) {
            // ---- A fragments (16-bit A 16x32 layout):
            // lane (lsel, half): row M=lsel, K runs [ks+8*half, +8) and
            // [ks+16+8*half, +8) -> two contiguous 16-byte bf16 runs in LDS.
            const __bf16* rx = ldsx + lsel * KC + ks + 8 * half;
            const __bf16* ry = ldsy + lsel * KC + ks + 8 * half;
            const v8bf axlo = *(const v8bf*)(rx);
            const v8bf axhi = *(const v8bf*)(rx + 16);
            const v8bf aylo = *(const v8bf*)(ry);
            const v8bf ayhi = *(const v8bf*)(ry + 16);
            const v16bf ax = __builtin_shufflevector(
                axlo, axhi, 0, 1, 2, 3, 4, 5, 6, 7, 8, 9, 10, 11, 12, 13, 14, 15);
            const v16bf ay = __builtin_shufflevector(
                aylo, ayhi, 0, 1, 2, 3, 4, 5, 6, 7, 8, 9, 10, 11, 12, 13, 14, 15);

            // ---- B fragments (16-bit B 32x16 layout): lane lsel holds
            // column N=lsel; K runs [ks+8*half, +8) and [ks+16+8*half, +8)
            // -> two 16-byte bf16 runs from the weight row (L2-resident).
            const __bf16* q0 = w0row + kc + ks + 8 * half;
            const __bf16* q1 = w1row + kc + ks + 8 * half;
            const v16bf b0 = __builtin_shufflevector(
                *(const v8bf*)(q0), *(const v8bf*)(q0 + 16), 0, 1, 2, 3, 4, 5, 6, 7, 8, 9, 10, 11, 12, 13, 14, 15);
            const v16bf b1 = __builtin_shufflevector(
                *(const v8bf*)(q1), *(const v8bf*)(q1 + 16), 0, 1, 2, 3, 4, 5, 6, 7, 8, 9, 10, 11, 12, 13, 14, 15);

            acc0 = __builtin_amdgcn_wmma_f32_16x16x32_bf16(
                false, ax, false, b0, (short)0, acc0, false, false);
            acc1 = __builtin_amdgcn_wmma_f32_16x16x32_bf16(
                false, ay, false, b1, (short)0, acc1, false, false);
        }
    }

    // ---- Fused epilogue. C/D layout: VGPR v, lanes 0-15 -> M=v, N=lane;
    // lanes 16-31 -> M=v+8, N=lane-16.
    const int   col = wave * 16 + lsel;
    const float A0  = a0[col];
    const float A1  = a1[col];
    const float Bb  = bias[col];

#pragma unroll
    for (int v = 0; v < 8; ++v) {
        const size_t m   = (size_t)(mrow0 + v + 8 * half);
        const float  o0  = acc0[v];
        const float  o1  = acc1[v];
        const size_t idx = m * OUTD + col;
        out0[idx] = o0;
        out1[idx] = o1;
        outf[idx] = o0 * A0 + o1 * A1 + Bb;
    }
}

// ---------------------------------------------------------------------------
// Launch
// Inputs (setup_inputs order): x, y, fc_weight, fc_bias, a_0, a_1
// Output: (out0, out1, output) concatenated flat -> 3 * 65536 * 128 f32
// ---------------------------------------------------------------------------
extern "C" void kernel_launch(void* const* d_in, const int* in_sizes, int n_in,
                              void* d_out, int out_size, void* d_ws, size_t ws_size,
                              hipStream_t stream) {
    (void)in_sizes; (void)n_in; (void)out_size; (void)ws_size;

    const float* x  = (const float*)d_in[0];
    const float* y  = (const float*)d_in[1];
    const float* W  = (const float*)d_in[2];
    const float* b  = (const float*)d_in[3];
    const float* a0 = (const float*)d_in[4];
    const float* a1 = (const float*)d_in[5];

    __bf16* Wn = (__bf16*)d_ws;                       // 256 * 1024 bf16 = 512 KB

    float* out0 = (float*)d_out;
    float* out1 = out0 + (size_t)BATCHN * OUTD;
    float* outf = out1 + (size_t)BATCHN * OUTD;

    own_norm_kernel<<<256, 256, 0, stream>>>(W, Wn);

    fused_ownnorm_gemm_kernel<<<BATCHN / MTILE, 256, 0, stream>>>(
        x, y, Wn, b, a0, a1, out0, out1, outf);
}
